// GAT_57354993271414
// MI455X (gfx1250) — matrix operand order repaired and checked by reading, not compiled, not run on hardware
//
#include <hip/hip_runtime.h>
#include <hip/hip_bf16.h>

typedef __attribute__((ext_vector_type(16))) _Float16 v16h;
typedef __attribute__((ext_vector_type(8)))  float    v8f;

#define N_NODES_C 262144
#define N_EDGES_C 2097152
#define C_C       64
#define FC_IN_C   2048
#define FC_HID_C  256
#define OUT_DIM_C 32
#define N_BATCH_C 8192

// ---------------------------------------------------------------------------
// Pack a row-major f32 weight matrix [K x N] into WMMA B-fragment order:
// fragment (kt, nt) = 32x16 tile; per lane 16 halves; lanes 0-15: K=kt*32+0..15
// (col = lane), lanes 16-31: K=kt*32+16..31 (col = lane-16). Element j <-> K+j.
// ---------------------------------------------------------------------------
__global__ void pack_b_f16(const float* __restrict__ W, _Float16* __restrict__ Bp,
                           int K, int N) {
  int idx = blockIdx.x * blockDim.x + threadIdx.x;
  if (idx >= K * N) return;
  int k = idx / N, n = idx - k * N;
  int kt = k >> 5, kk = k & 31;
  int nt = n >> 4, nn = n & 15;
  int Nt = N >> 4;
  int lane = nn + ((kk >> 4) << 4);
  int j = kk & 15;
  size_t dst = ((size_t)(kt * Nt + nt) * 32 + lane) * 16 + j;
  Bp[dst] = (_Float16)W[idx];
}

// ---------------------------------------------------------------------------
// Load one 16x32 f16 A fragment from a row-major f32 matrix (ld = K).
// Contiguous-k layout used here (matches pack_b_f16): lanes 0-15 (M=lane):  elems 0..15 = K 0..15
//             lanes 16-31 (M=lane-16): elems 0..15 = K 16..31
// ---------------------------------------------------------------------------
static __device__ __forceinline__ v16h load_a_frag_f32(const float* __restrict__ A,
                                                       int K, int row, int kb, int lane) {
  const int khalf = (lane >> 4) << 4;
  const float* p = A + (size_t)row * K + kb + khalf;
  float4 f0 = ((const float4*)p)[0];
  float4 f1 = ((const float4*)p)[1];
  float4 f2 = ((const float4*)p)[2];
  float4 f3 = ((const float4*)p)[3];
  float t[16] = {f0.x, f0.y, f0.z, f0.w, f1.x, f1.y, f1.z, f1.w,
                 f2.x, f2.y, f2.z, f2.w, f3.x, f3.y, f3.z, f3.w};
  v16h a;
#pragma unroll
  for (int i = 0; i < 16; ++i) a[i] = (_Float16)t[i];
  return a;
}

// ---------------------------------------------------------------------------
// Generic WMMA GEMM: Out[M x N] = A[M x K] (f32) * Bpacked (f16 fragments).
// Each wave computes a 16 x (NTW*16) tile. Optional bias + relu epilogue.
// Optional fused attention dots (NTW==4, full-row case): aS = h.attS, aD = h.attD.
// ---------------------------------------------------------------------------
template <int NTW>
__global__ __launch_bounds__(256) void wmma_gemm(
    const float* __restrict__ A, const _Float16* __restrict__ Bp,
    float* __restrict__ Out,
    const float* __restrict__ bias, int doRelu,
    const float* __restrict__ attS, const float* __restrict__ attD,
    float* __restrict__ aS, float* __restrict__ aD,
    int Mtiles, int K, int Nt) {
  const int lane = threadIdx.x & 31;
  const int wave = (blockIdx.x * blockDim.x + threadIdx.x) >> 5;
  const int ngroups = Nt / NTW;
  const int mtile = wave / ngroups;
  const int ng = wave - mtile * ngroups;
  if (mtile >= Mtiles) return;
  const int row0 = mtile << 4;
  const int n0 = ng * (NTW * 16);
  const int N = Nt << 4;
  const int Kt = K >> 5;

  v8f acc[NTW] = {};

  const int arow = row0 + (lane & 15);
  for (int kt = 0; kt < Kt; ++kt) {
    v16h a = load_a_frag_f32(A, K, arow, kt << 5, lane);
    const _Float16* bbase = Bp + ((size_t)kt * Nt + (n0 >> 4)) * 512 + lane * 16;
#pragma unroll
    for (int t = 0; t < NTW; ++t) {
      v16h b = *(const v16h*)(bbase + (size_t)t * 512);
      acc[t] = __builtin_amdgcn_wmma_f32_16x16x32_f16(
          false, a, false, b, (short)0, acc[t], false, false);
    }
  }

  // C layout: VGPR i -> (lanes 0-15: row row0+i, col n0+..+lane),
  //                     (lanes 16-31: row row0+8+i, col n0+..+lane-16)
  const int rbase = row0 + ((lane >> 4) << 3);
  const int cl = lane & 15;
#pragma unroll
  for (int t = 0; t < NTW; ++t) {
    const int col = n0 + t * 16 + cl;
    float bv = bias ? bias[col] : 0.0f;
#pragma unroll
    for (int i = 0; i < 8; ++i) {
      float v = acc[t][i] + bv;
      if (doRelu) v = fmaxf(v, 0.0f);
      Out[(size_t)(rbase + i) * N + col] = v;
    }
  }

  if constexpr (NTW == 4) {
    if (attS != nullptr) {  // fused per-row attention dot products (N == 64)
      float as[4], ad[4];
#pragma unroll
      for (int t = 0; t < 4; ++t) {
        as[t] = attS[t * 16 + cl];
        ad[t] = attD[t * 16 + cl];
      }
#pragma unroll
      for (int i = 0; i < 8; ++i) {
        float ps = acc[0][i] * as[0] + acc[1][i] * as[1] +
                   acc[2][i] * as[2] + acc[3][i] * as[3];
        float pd = acc[0][i] * ad[0] + acc[1][i] * ad[1] +
                   acc[2][i] * ad[2] + acc[3][i] * ad[3];
#pragma unroll
        for (int m = 8; m >= 1; m >>= 1) {
          ps += __shfl_xor(ps, m, 32);
          pd += __shfl_xor(pd, m, 32);
        }
        if (cl == 0) {
          aS[rbase + i] = ps;
          aD[rbase + i] = pd;
        }
      }
    }
  }
}

// ---------------------------------------------------------------------------
// Edge-wise segment softmax + aggregation (h is L2-resident: 64MB << 192MB L2)
// All float atomics use __hip_atomic_* (relaxed, agent scope) so they lower to
// single hardware global_atomic_{add_f32,max_i32,min_u32} ops, never CAS loops.
// ---------------------------------------------------------------------------
static __device__ __forceinline__ void atomAddF32(float* p, float v) {
  __hip_atomic_fetch_add(p, v, __ATOMIC_RELAXED, __HIP_MEMORY_SCOPE_AGENT);
}

static __device__ __forceinline__ void atomMaxF32(float* addr, float v) {
  // monotone int/uint encoding of IEEE754 ordering
  if (v >= 0.0f)
    __hip_atomic_fetch_max((int*)addr, __float_as_int(v),
                           __ATOMIC_RELAXED, __HIP_MEMORY_SCOPE_AGENT);
  else
    __hip_atomic_fetch_min((unsigned int*)addr, __float_as_uint(v),
                           __ATOMIC_RELAXED, __HIP_MEMORY_SCOPE_AGENT);
}

__global__ void fill_f32(float* __restrict__ p, float v, int n) {
  int i = blockIdx.x * blockDim.x + threadIdx.x;
  if (i < n) p[i] = v;
}

__global__ void edge_score(const int* __restrict__ src, const int* __restrict__ dst,
                           const float* __restrict__ aS, const float* __restrict__ aD,
                           float* __restrict__ e, float* __restrict__ m, int nE) {
  int i = blockIdx.x * blockDim.x + threadIdx.x;
  if (i >= nE) return;
  float v = aS[src[i]] + aD[dst[i]];
  v = v > 0.0f ? v : 0.2f * v;      // leaky_relu, slope 0.2
  e[i] = v;
  atomMaxF32(&m[dst[i]], v);
}

__global__ void edge_exp(const int* __restrict__ dst, float* __restrict__ e,
                         const float* __restrict__ m, float* __restrict__ s, int nE) {
  int i = blockIdx.x * blockDim.x + threadIdx.x;
  if (i >= nE) return;
  int d = dst[i];
  float ex = __expf(e[i] - m[d]);
  e[i] = ex;
  atomAddF32(&s[d], ex);
}

__global__ void edge_aggregate(const int* __restrict__ src, const int* __restrict__ dst,
                               const float* __restrict__ e, const float* __restrict__ s,
                               const float* __restrict__ h, float* __restrict__ agg,
                               int nE) {
  int t = blockIdx.x * blockDim.x + threadIdx.x;   // nE*64 = 134M < 2^31
  int i = t >> 6;
  int c = t & 63;
  if (i >= nE) return;
  int d = dst[i];
  float alpha = e[i] / s[d];
  atomAddF32(&agg[(size_t)d * 64 + c], alpha * h[(size_t)src[i] * 64 + c]);
}

__global__ void add_bias64(float* __restrict__ x, const float* __restrict__ b, int n) {
  int i = blockIdx.x * blockDim.x + threadIdx.x;
  if (i < n) x[i] += b[i & 63];
}

// ---------------------------------------------------------------------------
extern "C" void kernel_launch(void* const* d_in, const int* in_sizes, int n_in,
                              void* d_out, int out_size, void* d_ws, size_t ws_size,
                              hipStream_t stream) {
  const float* x        = (const float*)d_in[0];
  const int*   ei       = (const int*)d_in[1];
  const float* W1       = (const float*)d_in[2];
  const float* att_src1 = (const float*)d_in[3];
  const float* att_dst1 = (const float*)d_in[4];
  const float* b1       = (const float*)d_in[5];
  const float* W2       = (const float*)d_in[6];
  const float* att_src2 = (const float*)d_in[7];
  const float* att_dst2 = (const float*)d_in[8];
  const float* b2       = (const float*)d_in[9];
  const float* Wf1      = (const float*)d_in[10];
  const float* bf1      = (const float*)d_in[11];
  const float* Wf2      = (const float*)d_in[12];
  const float* bf2      = (const float*)d_in[13];
  const int* src = ei;
  const int* dst = ei + N_EDGES_C;

  // workspace carve-up (256B aligned)
  char* ws = (char*)d_ws;
  size_t off = 0;
  auto carve = [&](size_t bytes) {
    void* p = ws + off;
    off += (bytes + 255) & ~(size_t)255;
    return p;
  };
  float*    h    = (float*)carve((size_t)N_NODES_C * 64 * 4);   // x@W scratch (per layer)
  float*    xb   = (float*)carve((size_t)N_NODES_C * 64 * 4);   // agg out / next input / g
  float*    e    = (float*)carve((size_t)N_EDGES_C * 4);
  float*    aS   = (float*)carve((size_t)N_NODES_C * 4);
  float*    aD   = (float*)carve((size_t)N_NODES_C * 4);
  float*    m    = (float*)carve((size_t)N_NODES_C * 4);
  float*    s    = (float*)carve((size_t)N_NODES_C * 4);
  float*    f1o  = (float*)carve((size_t)N_BATCH_C * FC_HID_C * 4);
  _Float16* Bp1  = (_Float16*)carve((size_t)64 * 64 * 2);
  _Float16* Bp2  = (_Float16*)carve((size_t)64 * 64 * 2);
  _Float16* Bpf1 = (_Float16*)carve((size_t)FC_IN_C * FC_HID_C * 2);
  _Float16* Bpf2 = (_Float16*)carve((size_t)FC_HID_C * OUT_DIM_C * 2);

  const int T = 256;
  // 0) pack weights to WMMA f16 B-fragment order
  pack_b_f16<<<(64 * 64 + T - 1) / T, T, 0, stream>>>(W1, Bp1, 64, 64);
  pack_b_f16<<<(64 * 64 + T - 1) / T, T, 0, stream>>>(W2, Bp2, 64, 64);
  pack_b_f16<<<(FC_IN_C * FC_HID_C + T - 1) / T, T, 0, stream>>>(Wf1, Bpf1, FC_IN_C, FC_HID_C);
  pack_b_f16<<<(FC_HID_C * OUT_DIM_C + T - 1) / T, T, 0, stream>>>(Wf2, Bpf2, FC_HID_C, OUT_DIM_C);

  const int Mtiles_nodes = N_NODES_C / 16;             // 16384 waves, 1 ngroup
  const int gemm_nodes_blocks = (Mtiles_nodes * 32) / T;
  const int nodeElems = N_NODES_C * 64;
  const int eBlocks = (N_EDGES_C + T - 1) / T;
  const int aggBlocks = (N_EDGES_C * 64) / T;

  for (int layer = 0; layer < 2; ++layer) {
    const float*    Ain  = layer == 0 ? x : xb;
    const _Float16* Bp   = layer == 0 ? Bp1 : Bp2;
    const float*    atS  = layer == 0 ? att_src1 : att_src2;
    const float*    atD  = layer == 0 ? att_dst1 : att_dst2;
    const float*    bias = layer == 0 ? b1 : b2;

    // h = Ain @ W  (+ fused a_s = h.att_src, a_d = h.att_dst)
    wmma_gemm<4><<<gemm_nodes_blocks, T, 0, stream>>>(
        Ain, Bp, h, nullptr, 0, atS, atD, aS, aD, Mtiles_nodes, 64, 4);

    // segment softmax state (xb free after GEMM consumed it)
    fill_f32<<<(N_NODES_C + T - 1) / T, T, 0, stream>>>(m, -__builtin_inff(), N_NODES_C);
    fill_f32<<<(N_NODES_C + T - 1) / T, T, 0, stream>>>(s, 0.0f, N_NODES_C);
    fill_f32<<<(nodeElems + T - 1) / T, T, 0, stream>>>(xb, 0.0f, nodeElems);

    edge_score<<<eBlocks, T, 0, stream>>>(src, dst, aS, aD, e, m, N_EDGES_C);
    edge_exp<<<eBlocks, T, 0, stream>>>(dst, e, m, s, N_EDGES_C);
    edge_aggregate<<<aggBlocks, T, 0, stream>>>(src, dst, e, s, h, xb, N_EDGES_C);
    add_bias64<<<(nodeElems + T - 1) / T, T, 0, stream>>>(xb, bias, nodeElems);
  }

  // FC head: xb viewed as g [8192 x 2048]
  // fc1: Mtiles=512, Nt=16, ngroups=4 -> 2048 waves
  wmma_gemm<4><<<(512 * 4 * 32) / T, T, 0, stream>>>(
      xb, Bpf1, f1o, bf1, 1, nullptr, nullptr, nullptr, nullptr, 512, FC_IN_C, 16);
  // fc2: Mtiles=512, Nt=2, NTW=2 -> 512 waves
  wmma_gemm<2><<<(512 * 32) / T, T, 0, stream>>>(
      f1o, Bpf2, (float*)d_out, bf2, 0, nullptr, nullptr, nullptr, nullptr, 512, FC_HID_C, 2);
}
